// RNN_6081673691644
// MI455X (gfx1250) — matrix operand order repaired and checked by reading, not compiled, not run on hardware
//
#include <hip/hip_runtime.h>
#include <hip/hip_bf16.h>

// ---------------------------------------------------------------------------
// RNN (2-layer tanh Elman RNN + tanh output projection) on gfx1250 via
// v_wmma_f32_16x16x32_bf16. FP32 inputs are converted once to bf16 in the
// workspace; all GEMMs run on the WMMA pipe with f32 accumulation.
//
// Round 2 changes:
//  - Recurrent chain pipelined across layers: phase p = {layer0[t=p],
//    layer1[t=p-1]} in ONE kernel launch -> 71 launches instead of 140.
//  - Output projection blocked 2 M-tiles x 5 N-tiles per wave (625 = 125*5),
//    cutting VMEM issue per WMMA from 4 loads to 1.4.
//  - Recurrent K-loop split into A1/A2 halves (no per-iter pointer select).
// ---------------------------------------------------------------------------

#define SEQ_LEN 70
#define BATCH   64
#define VOCAB   10000
#define EMB     512
#define HID     512
#define HB      (BATCH * HID)   // 32768

typedef __attribute__((ext_vector_type(16))) __bf16 v16bf;
typedef __attribute__((ext_vector_type(8)))  float  v8f;

union FragU {
  uint4 u[2];
  v16bf v;
};

__device__ __forceinline__ unsigned short f2bf(float f) {
  unsigned int u = __float_as_uint(f);
  u += 0x7fffu + ((u >> 16) & 1u);   // round-to-nearest-even
  return (unsigned short)(u >> 16);
}
__device__ __forceinline__ float bf2f(unsigned short b) {
  return __uint_as_float(((unsigned int)b) << 16);
}

__device__ __forceinline__ v8f wmma_bf16(const FragU& a, const FragU& b, v8f c) {
  return __builtin_amdgcn_wmma_f32_16x16x32_bf16(
      false, a.v, false, b.v, (short)0, c, false, false);
}

// ---------------------------------------------------------------- converts
__global__ void k_f32_to_bf16(const float* __restrict__ src,
                              unsigned short* __restrict__ dst, int n) {
  int i = blockIdx.x * blockDim.x + threadIdx.x;
  if (i < n) dst[i] = f2bf(src[i]);
}

// Embedding gather + convert: X[(t*B+b)*E + e] = bf16(table[tok[t*B+b]*E + e])
__global__ void k_embed(const int* __restrict__ tok,
                        const float* __restrict__ table,
                        unsigned short* __restrict__ X, int n) {
  int i = blockIdx.x * blockDim.x + threadIdx.x;
  if (i >= n) return;
  int row = i >> 9;      // / EMB (=512)
  int e   = i & 511;
  int id  = tok[row];
  X[i] = f2bf(table[id * EMB + e]);
}

// ------------------------------------------------------- recurrent layer tile
// One wave computes one 16x16 tile of:
//   Out(64x512) = tanh( [A1 | A2](64x1024) @ W^T(1024x512) + bias )
// K split: K 0..511 from A1, K 512..1023 from A2 (no per-iter select).
__device__ __forceinline__ void layer_tile(
    const unsigned short* __restrict__ A1,   // 64x512 bf16 (k <  512)
    const unsigned short* __restrict__ A2,   // 64x512 bf16 (k >= 512)
    const unsigned short* __restrict__ W,    // 512x1024 bf16 row-major
    const float* __restrict__ bias,          // 512 fp32
    unsigned short* __restrict__ Out,        // 64x512 bf16
    int tile, int lane)
{
  const int laneLo = lane & 15;
  const int hi     = lane >> 4;                 // lane half
  const int tileN  = tile & 31;                 // 32 N-tiles
  const int tileM  = tile >> 5;                 // 4  M-tiles
  const int m0 = tileM * 16, n0 = tileN * 16;

  const unsigned short* Ar1  = A1 + (m0 + laneLo) * 512;
  const unsigned short* Ar2  = A2 + (m0 + laneLo) * 512;
  const unsigned short* Brow = W  + (n0 + laneLo) * 1024;

  v8f acc = {};
#pragma unroll 4
  for (int kt = 0; kt < 16; ++kt) {             // K 0..511 (A1 half)
    const int kb = kt * 32;
    FragU a, b;
    // A layout: lanes 0-15 hold K {0..7,16..23}; lanes 16-31 hold {8..15,24..31}
    a.u[0] = *(const uint4*)(Ar1 + kb + hi * 8);
    a.u[1] = *(const uint4*)(Ar1 + kb + hi * 8 + 16);
    // B layout: lanes 0-15 hold K {0..7,16..23}; lanes 16-31 hold {8..15,24..31}
    b.u[0] = *(const uint4*)(Brow + kb + hi * 8);
    b.u[1] = *(const uint4*)(Brow + kb + hi * 8 + 16);
    acc = wmma_bf16(a, b, acc);
  }
#pragma unroll 4
  for (int kt = 0; kt < 16; ++kt) {             // K 512..1023 (A2 half)
    const int kb = kt * 32;
    FragU a, b;
    a.u[0] = *(const uint4*)(Ar2 + kb + hi * 8);
    a.u[1] = *(const uint4*)(Ar2 + kb + hi * 8 + 16);
    b.u[0] = *(const uint4*)(Brow + 512 + kb + hi * 8);
    b.u[1] = *(const uint4*)(Brow + 512 + kb + hi * 8 + 16);
    acc = wmma_bf16(a, b, acc);
  }

  const int   col = n0 + laneLo;
  const float bv  = bias[col];
#pragma unroll
  for (int r = 0; r < 8; ++r) {
    const int row = m0 + r + hi * 8;            // C/D layout
    Out[row * 512 + col] = f2bf(tanhf(acc[r] + bv));
  }
}

// ------------------------------------------------------ pipelined chain phase
// Phase p: blocks 0..31  -> layer0 at t = p       (if p < SEQ_LEN)
//          blocks 32..63 -> layer1 at t = p - 1   (if p >= 1)
// Both are independent within a phase: layer1[p-1] consumes layer0 output of
// phase p-1 (slot p&1) while layer0[p] reads slot p&1 / writes slot (p+1)&1.
__global__ __launch_bounds__(128) void k_chain_phase(
    const unsigned short* __restrict__ X,    // 70*64*512 bf16 embeddings
    unsigned short* __restrict__ h0b,        // 2 * 64*512 bf16 ping-pong
    unsigned short* __restrict__ H1,         // 70*64*512 bf16 layer-1 outputs
    const unsigned short* __restrict__ h1i,  // 64*512 bf16 initial h1
    const unsigned short* __restrict__ W0b,
    const unsigned short* __restrict__ W1b,
    const float* __restrict__ b0,
    const float* __restrict__ b1,
    int p)
{
  const int lane = threadIdx.x & 31;
  const int wave = threadIdx.x >> 5;

  if (blockIdx.x < 32) {                       // ---- layer 0 at t = p
    if (p >= SEQ_LEN) return;
    const int t = p;
    const unsigned short* xt     = X + t * (BATCH * EMB);
    const unsigned short* h0prev = h0b + (t & 1) * HB;
    unsigned short*       h0next = h0b + ((t + 1) & 1) * HB;
    layer_tile(xt, h0prev, W0b, b0, h0next, blockIdx.x * 4 + wave, lane);
  } else {                                     // ---- layer 1 at t = p - 1
    if (p < 1) return;
    const int t = p - 1;
    const unsigned short* a1     = h0b + ((t + 1) & 1) * HB;   // layer0 out @ t
    const unsigned short* h1prev = (t == 0) ? h1i : (H1 + (t - 1) * HB);
    layer_tile(a1, h1prev, W1b, b1, H1 + t * HB,
               (blockIdx.x - 32) * 4 + wave, lane);
  }
}

// ----------------------------------------------------- output projection GEMM
// logits(4480 x 10000, f32) = tanh( H1(4480x512,bf16) @ Wout^T + bout )
// Wave blocking: 2 M-tiles x 5 N-tiles (625 N-tiles = 125 * 5 exactly).
// grid = (125, 35), block = 128 (4 waves -> 8 M-tiles per block, 280 = 35*8).
// Per K-step: 4 A-loads + 10 B-loads feed 10 WMMAs (1.4 loads/WMMA).
__global__ __launch_bounds__(128) void k_out_gemm(
    const unsigned short* __restrict__ H1,   // 4480 x 512 bf16
    const unsigned short* __restrict__ Wb,   // 10000 x 512 bf16 row-major
    const float* __restrict__ bias,          // 10000 fp32
    float* __restrict__ out)                 // 4480 x 10000 fp32
{
  const int wave   = threadIdx.x >> 5;
  const int lane   = threadIdx.x & 31;
  const int laneLo = lane & 15;
  const int hi     = lane >> 4;

  const int mt0 = (blockIdx.y * 4 + wave) * 2;  // first of 2 M-tiles
  const int nt0 = blockIdx.x * 5;               // first of 5 N-tiles

  const unsigned short* Arow0 = H1 + (mt0 * 16 + laneLo) * 512;
  const unsigned short* Arow1 = Arow0 + 16 * 512;
  const unsigned short* Brow[5];
#pragma unroll
  for (int j = 0; j < 5; ++j)
    Brow[j] = Wb + ((nt0 + j) * 16 + laneLo) * 512;

  v8f acc[2][5] = {};
#pragma unroll 2
  for (int kt = 0; kt < 16; ++kt) {
    const int kb = kt * 32;
    FragU a0, a1, b[5];
    a0.u[0] = *(const uint4*)(Arow0 + kb + hi * 8);
    a0.u[1] = *(const uint4*)(Arow0 + kb + hi * 8 + 16);
    a1.u[0] = *(const uint4*)(Arow1 + kb + hi * 8);
    a1.u[1] = *(const uint4*)(Arow1 + kb + hi * 8 + 16);
#pragma unroll
    for (int j = 0; j < 5; ++j) {
      b[j].u[0] = *(const uint4*)(Brow[j] + kb + hi * 8);
      b[j].u[1] = *(const uint4*)(Brow[j] + kb + hi * 8 + 16);
    }
#pragma unroll
    for (int j = 0; j < 5; ++j) {
      acc[0][j] = wmma_bf16(a0, b[j], acc[0][j]);
      acc[1][j] = wmma_bf16(a1, b[j], acc[1][j]);
    }
  }

#pragma unroll
  for (int i = 0; i < 2; ++i) {
#pragma unroll
    for (int j = 0; j < 5; ++j) {
      const int   col = (nt0 + j) * 16 + laneLo;
      const float bv  = bias[col];
#pragma unroll
      for (int r = 0; r < 8; ++r) {
        const int row = (mt0 + i) * 16 + r + hi * 8;
        out[(size_t)row * VOCAB + col] = tanhf(acc[i][j][r] + bv);
      }
    }
  }
}

// ----------------------------------------------------------- h_final tail
__global__ void k_hfinal(const unsigned short* __restrict__ h0,
                         const unsigned short* __restrict__ h1,
                         float* __restrict__ dst) {
  int i = blockIdx.x * blockDim.x + threadIdx.x;
  if (i < HB)          dst[i] = bf2f(h0[i]);
  else if (i < 2 * HB) dst[i] = bf2f(h1[i - HB]);
}

// ---------------------------------------------------------------------------
extern "C" void kernel_launch(void* const* d_in, const int* in_sizes, int n_in,
                              void* d_out, int out_size, void* d_ws, size_t ws_size,
                              hipStream_t stream) {
  (void)in_sizes; (void)n_in; (void)out_size; (void)ws_size;

  const int*   inputs = (const int*)  d_in[0];   // (70, 64) int32
  const float* hidden = (const float*)d_in[1];   // (2, 64, 512) f32
  const float* embT   = (const float*)d_in[2];   // (10000, 512) f32
  const float* W0     = (const float*)d_in[3];   // (512, 1024)
  const float* b0     = (const float*)d_in[4];   // (512,)
  const float* W1     = (const float*)d_in[5];   // (512, 1024)
  const float* b1     = (const float*)d_in[6];   // (512,)
  const float* Wout   = (const float*)d_in[7];   // (10000, 512)
  const float* bout   = (const float*)d_in[8];   // (10000,)
  float* out = (float*)d_out;

  // Workspace layout (byte offsets, 256B-aligned):
  char* ws = (char*)d_ws;
  unsigned short* W0b = (unsigned short*)(ws + 0);          // 512*1024  (1,048,576 B)
  unsigned short* W1b = (unsigned short*)(ws + 1048576);    // 512*1024
  unsigned short* Wob = (unsigned short*)(ws + 2097152);    // 10000*512 (10,240,000 B)
  unsigned short* X   = (unsigned short*)(ws + 12337152);   // 70*64*512 (4,587,520 B)
  unsigned short* H1  = (unsigned short*)(ws + 16924672);   // 70*64*512
  unsigned short* h0b = (unsigned short*)(ws + 21512192);   // 2 * 64*512 ping-pong
  unsigned short* h1i = (unsigned short*)(ws + 21643264);   // 64*512 initial h1

  // 1) one-time f32 -> bf16 conversions (weights + initial hidden state)
  auto cvt = [&](const float* s, unsigned short* d, int n) {
    k_f32_to_bf16<<<(n + 255) / 256, 256, 0, stream>>>(s, d, n);
  };
  cvt(W0, W0b, 512 * 1024);
  cvt(W1, W1b, 512 * 1024);
  cvt(Wout, Wob, VOCAB * HID);
  cvt(hidden,      h0b, HB);   // layer-0 init -> ping slot 0
  cvt(hidden + HB, h1i, HB);   // layer-1 init

  // 2) embedding gather (whole sequence)
  const int nemb = SEQ_LEN * BATCH * EMB;
  k_embed<<<(nemb + 255) / 256, 256, 0, stream>>>(inputs, embT, X, nemb);

  // 3) software-pipelined recurrent chain: 71 phases (layer0[p] || layer1[p-1])
  for (int p = 0; p <= SEQ_LEN; ++p) {
    k_chain_phase<<<64, 128, 0, stream>>>(X, h0b, H1, h1i, W0b, W1b, b0, b1, p);
  }

  // 4) bulk output projection: (70*64) x 10000, K=512, fused tanh+bias
  dim3 grid(125, 35);
  k_out_gemm<<<grid, 128, 0, stream>>>(H1, Wob, bout, out);

  // 5) h_final: after t=69, h0 lives in slot ((69+1)&1)=0; h1 is last H1 block
  k_hfinal<<<(2 * HB + 255) / 256, 256, 0, stream>>>(
      h0b, H1 + (SEQ_LEN - 1) * HB, out + (size_t)SEQ_LEN * BATCH * VOCAB);
}
